// TransformerBlock_59631325938238
// MI455X (gfx1250) — compile-verified, matrix operand order checked by reading, not run on hardware
//
#include <hip/hip_runtime.h>
#include <hip/hip_bf16.h>
#include <stdint.h>

// ---------------------------------------------------------------------------
// Types & helpers
// ---------------------------------------------------------------------------
typedef __attribute__((ext_vector_type(16))) __bf16 bhalf16;
typedef __attribute__((ext_vector_type(8)))  float  float8;
typedef __attribute__((ext_vector_type(4)))  unsigned int uint32x4;
typedef __attribute__((ext_vector_type(8)))  int int32x8;
typedef __attribute__((ext_vector_type(4)))  int int32x4;

union FragAB {
  bhalf16 v;
  uint4   q[2];
};

__device__ __forceinline__ unsigned short f2bf(float x) {
  union { float f; unsigned u; } v; v.f = x;
  unsigned r = v.u + 0x7FFFu + ((v.u >> 16) & 1u);   // round-to-nearest-even
  return (unsigned short)(r >> 16);
}
__device__ __forceinline__ float bf2f(unsigned short h) {
  union { unsigned u; float f; } v; v.u = ((unsigned)h) << 16;
  return v.f;
}

// A/B fragment load per CDNA5 16-bit layout:
// lanes 0-15: row m=lane,    K = kofs+{0..7} then kofs+{16..23}
// lanes 16-31: row m=lane-16, K = kofs+{8..15} then kofs+{24..31}
__device__ __forceinline__ void load_frag(FragAB& f, const unsigned short* base,
                                          int stride, int kofs, int lane) {
  const int g = (lane >> 4) & 1;
  const int m = lane & 15;
  const unsigned short* p = base + m * stride + kofs + g * 8;
  f.q[0] = *(const uint4*)(p);
  f.q[1] = *(const uint4*)(p + 16);
}

__device__ __forceinline__ float8 wmma_bf16(const FragAB& a, const FragAB& b, float8 c) {
  return __builtin_amdgcn_wmma_f32_16x16x32_bf16(false, a.v, false, b.v,
                                                 (short)0, c, false, false);
}

// Generic LDS pointer -> raw LDS byte offset (LDS aperture: addr[31:0]).
__device__ __forceinline__ unsigned lds_off(const void* p) {
  return (unsigned)(unsigned long long)p;
}

// ---------------------------------------------------------------------------
// TDM: DMA one bf16 tile [rows x 32] (row stride = stride_elems in global) into
// LDS with 16B padding after each 64B row (-> padded LDS stride of 40 elems).
// D# packing per CDNA5 ISA ch.8.3-8.5. Issued by one wave; TENSORcnt tracked.
// This toolchain's builtin takes 6 args (g0, g1, g2, g3, g_extra, cpol).
// ---------------------------------------------------------------------------
__device__ __forceinline__ void tdm_load_tile32(unsigned lds_byte_off,
                                                const void* gaddr,
                                                unsigned rows,
                                                unsigned stride_elems)
{
  uint32x4 g0;
  g0[0] = 1u;                                   // count=1, user-mode, no gather
  g0[1] = lds_byte_off;                         // lds_addr
  unsigned long long ga = (unsigned long long)gaddr;
  g0[2] = (unsigned)ga;                         // global_addr[31:0]
  g0[3] = (unsigned)(ga >> 32) | (2u << 30);    // global_addr[56:32] | type=2

  const unsigned td = 1u << 28;                 // huge tensor dims: tile never OOB
  int32x8 g1;
  g1[0] = (int)((1u << 16)                      // data_size = 2 bytes
              | (1u << 20)                      // pad_enable
              | (3u << 22)                      // pad_interval: 16 DWORDs (64B)
              | (3u << 25));                    // pad_amount: 4 DWORDs (16B)
  g1[1] = (int)((td & 0xFFFFu) << 16);          // tensor_dim0[15:0]
  g1[2] = (int)((td >> 16) | ((td & 0xFFFFu) << 16));   // d0 hi | d1 lo
  g1[3] = (int)((td >> 16) | (32u << 16));      // d1 hi | tile_dim0 = 32 elems
  g1[4] = (int)(rows & 0xFFFFu);                // tile_dim1 = rows, tile_dim2 = 0
  g1[5] = (int)stride_elems;                    // tensor_dim0_stride[31:0]
  g1[6] = 0;                                    // stride hi | dim1_stride lo
  g1[7] = 0;
  const int32x4 z4 = {0, 0, 0, 0};              // groups 2/3 unused (2D tile)
  const int32x8 z8 = {0, 0, 0, 0, 0, 0, 0, 0};
  __builtin_amdgcn_tensor_load_to_lds(g0, g1, z4, z4, z8, 0);
}

// ---------------------------------------------------------------------------
// GEMM: C[M,N] = A[M,K](bf16) * Bt[N,K](bf16)^T  + epilogue
// epilogue: v = acc (+bias[n]) (*rowscale[m*rs_stride]) (+resid[m,n])
//           out_bf16 ? store bf16 : (accumulate ? C+=v : C=v)
// Tiles: 128x128x32, 256 threads (8 waves, 2x4 wave grid, 64x32 per wave).
// A/B tiles arrive via TDM (double-buffered LDS, DMA overlaps WMMA compute).
// ---------------------------------------------------------------------------
#define BK  32
#define LDT 40   // padded LDS row stride (elems): 80B = 20 banks, conflict-free

__global__ __launch_bounds__(256) void gemm_bf16_kernel(
    const unsigned short* __restrict__ A, const unsigned short* __restrict__ Bt,
    const float* __restrict__ bias, const float* __restrict__ resid,
    const float* __restrict__ rowscale, int rs_stride,
    void* __restrict__ Cout, int out_bf16, int accumulate,
    int M, int N, int K)
{
  __shared__ __attribute__((aligned(16))) unsigned short As[2][128 * LDT];
  __shared__ __attribute__((aligned(16))) unsigned short Bs[2][128 * LDT];

  const int tid  = threadIdx.x;
  const int lane = tid & 31;
  const int wave = tid >> 5;
  const int wr   = wave >> 2;     // 0..1 -> 64-row slab
  const int wc   = wave & 3;      // 0..3 -> 32-col slab
  const int m0   = blockIdx.y * 128;
  const int n0   = blockIdx.x * 128;

  float8 acc[4][2];
#pragma unroll
  for (int i = 0; i < 4; ++i)
#pragma unroll
    for (int j = 0; j < 2; ++j)
#pragma unroll
      for (int r = 0; r < 8; ++r) acc[i][j][r] = 0.0f;

  const unsigned ldsA0 = lds_off(&As[0][0]);
  const unsigned ldsA1 = lds_off(&As[1][0]);
  const unsigned ldsB0 = lds_off(&Bs[0][0]);
  const unsigned ldsB1 = lds_off(&Bs[1][0]);

  // Prologue: kick off tile 0 (wave0 -> A, wave1 -> B)
  if (wave == 0)      tdm_load_tile32(ldsA0, A  + (size_t)m0 * K, 128, (unsigned)K);
  else if (wave == 1) tdm_load_tile32(ldsB0, Bt + (size_t)n0 * K, 128, (unsigned)K);

  int cur = 0;
  for (int k0 = 0; k0 < K; k0 += BK) {
    __syncthreads();   // everyone done computing from buf[cur^1] -> safe to refill
    const int nxt = k0 + BK;
    if (nxt < K) {
      if (wave == 0)
        tdm_load_tile32(cur ? ldsA0 : ldsA1, A  + (size_t)m0 * K + nxt, 128, (unsigned)K);
      else if (wave == 1)
        tdm_load_tile32(cur ? ldsB0 : ldsB1, Bt + (size_t)n0 * K + nxt, 128, (unsigned)K);
      if (wave < 2) __builtin_amdgcn_s_wait_tensorcnt(1);  // tile k0 landed (in-order)
    } else {
      if (wave < 2) __builtin_amdgcn_s_wait_tensorcnt(0);
    }
    __syncthreads();   // tile k0 visible to all waves

    const unsigned short* Ab = As[cur];
    const unsigned short* Bb = Bs[cur];
    FragAB af[4], bf[2];
#pragma unroll
    for (int i = 0; i < 4; ++i)
      load_frag(af[i], &Ab[(wr * 64 + i * 16) * LDT], LDT, 0, lane);
#pragma unroll
    for (int j = 0; j < 2; ++j)
      load_frag(bf[j], &Bb[(wc * 32 + j * 16) * LDT], LDT, 0, lane);
#pragma unroll
    for (int i = 0; i < 4; ++i)
#pragma unroll
      for (int j = 0; j < 2; ++j)
        acc[i][j] = wmma_bf16(af[i], bf[j], acc[i][j]);

    cur ^= 1;
  }

  const int g  = (lane >> 4) & 1;
  const int ln = lane & 15;
#pragma unroll
  for (int i = 0; i < 4; ++i)
#pragma unroll
    for (int j = 0; j < 2; ++j)
#pragma unroll
      for (int r = 0; r < 8; ++r) {
        const int row = m0 + wr * 64 + i * 16 + r + g * 8;
        const int col = n0 + wc * 32 + j * 16 + ln;
        float v = acc[i][j][r];
        if (bias)     v += bias[col];
        if (rowscale) v *= rowscale[(size_t)row * rs_stride];
        if (resid)    v += resid[(size_t)row * N + col];
        const size_t idx = (size_t)row * N + col;
        if (out_bf16) {
          ((unsigned short*)Cout)[idx] = f2bf(v);
        } else {
          float* Cf = (float*)Cout;
          float o = v;
          if (accumulate) o += Cf[idx];
          Cf[idx] = o;
        }
      }
}

// ---------------------------------------------------------------------------
// Flash attention: per block = (q-tile of 64, head). 4 waves, 16 Q rows/wave.
// Q,K: [B*H, S, 64] bf16 ; Vt: [B*H, 64, S] bf16 ; O: [B*S, H*64] bf16
// ---------------------------------------------------------------------------
#define LDA 72   // padded stride for 64-wide attn tiles (144B rows, no conflict)

__global__ __launch_bounds__(128) void attn_kernel(
    const unsigned short* __restrict__ Q, const unsigned short* __restrict__ Kg,
    const unsigned short* __restrict__ Vt, unsigned short* __restrict__ O,
    int S, int H, float scale)
{
  const int bh   = blockIdx.y;
  const int b    = bh / H, h = bh % H;
  const int q0   = blockIdx.x * 64;
  const int tid  = threadIdx.x;
  const int lane = tid & 31;
  const int wave = tid >> 5;
  const int g    = (lane >> 4) & 1;
  const int ln   = lane & 15;

  __shared__ __attribute__((aligned(16))) unsigned short Qs[64 * LDA];
  __shared__ __attribute__((aligned(16))) unsigned short Ks[64 * LDA];
  __shared__ __attribute__((aligned(16))) unsigned short Vs[64 * LDA]; // [d][t]
  __shared__ __attribute__((aligned(16))) unsigned short Ps[64 * LDA]; // per-wave P

  const int lrow  = tid >> 1;   // 0..63
  const int lhalf = tid & 1;    // 0..1 (32-elem half)

  { // load Q tile once
    const uint4* src = (const uint4*)(Q + ((size_t)bh * S + q0 + lrow) * 64 + lhalf * 32);
    uint4* dst = (uint4*)(&Qs[lrow * LDA + lhalf * 32]);
    dst[0] = src[0]; dst[1] = src[1]; dst[2] = src[2]; dst[3] = src[3];
  }
  __syncthreads();
  FragAB qf[2];
  load_frag(qf[0], &Qs[(wave * 16) * LDA], LDA, 0,  lane);
  load_frag(qf[1], &Qs[(wave * 16) * LDA], LDA, 32, lane);

  float8 oacc[4];
  float  m_r[8], l_r[8];
#pragma unroll
  for (int nt = 0; nt < 4; ++nt)
#pragma unroll
    for (int r = 0; r < 8; ++r) oacc[nt][r] = 0.0f;
#pragma unroll
  for (int r = 0; r < 8; ++r) { m_r[r] = -3.0e38f; l_r[r] = 0.0f; }

  unsigned short* prow = &Ps[(wave * 16) * LDA];
  const int nT = S / 64;

  for (int j = 0; j < nT; ++j) {
    __syncthreads();
    { // K tile: [t][d]
      const uint4* src = (const uint4*)(Kg + ((size_t)bh * S + j * 64 + lrow) * 64 + lhalf * 32);
      uint4* dst = (uint4*)(&Ks[lrow * LDA + lhalf * 32]);
      dst[0] = src[0]; dst[1] = src[1]; dst[2] = src[2]; dst[3] = src[3];
    }
    { // Vt tile: [d][t]
      const uint4* src = (const uint4*)(Vt + ((size_t)bh * 64 + lrow) * S + j * 64 + lhalf * 32);
      uint4* dst = (uint4*)(&Vs[lrow * LDA + lhalf * 32]);
      dst[0] = src[0]; dst[1] = src[1]; dst[2] = src[2]; dst[3] = src[3];
    }
    __syncthreads();

    // S = Q * K^T  (B-frag = K rows, contiguous d)
    float8 sacc[4];
#pragma unroll
    for (int nt = 0; nt < 4; ++nt)
#pragma unroll
      for (int r = 0; r < 8; ++r) sacc[nt][r] = 0.0f;
#pragma unroll
    for (int kb = 0; kb < 2; ++kb)
#pragma unroll
      for (int nt = 0; nt < 4; ++nt) {
        FragAB kf;
        load_frag(kf, &Ks[(nt * 16) * LDA], LDA, kb * 32, lane);
        sacc[nt] = wmma_bf16(qf[kb], kf, sacc[nt]);
      }
#pragma unroll
    for (int nt = 0; nt < 4; ++nt)
#pragma unroll
      for (int r = 0; r < 8; ++r) sacc[nt][r] *= scale;

    // online softmax (rows striped: row = r + g*8, cols across 16 lanes x 4 tiles)
    float mn[8], al[8];
#pragma unroll
    for (int r = 0; r < 8; ++r) {
      float v = fmaxf(fmaxf(sacc[0][r], sacc[1][r]), fmaxf(sacc[2][r], sacc[3][r]));
#pragma unroll
      for (int off = 1; off < 16; off <<= 1) v = fmaxf(v, __shfl_xor(v, off, 32));
      mn[r] = fmaxf(m_r[r], v);
      al[r] = __expf(m_r[r] - mn[r]);
      m_r[r] = mn[r];
    }
    float rs[8];
#pragma unroll
    for (int r = 0; r < 8; ++r) rs[r] = 0.0f;
#pragma unroll
    for (int nt = 0; nt < 4; ++nt)
#pragma unroll
      for (int r = 0; r < 8; ++r) {
        float p = __expf(sacc[nt][r] - mn[r]);
        sacc[nt][r] = p;
        rs[r] += p;
      }
#pragma unroll
    for (int r = 0; r < 8; ++r) {
      float v = rs[r];
#pragma unroll
      for (int off = 1; off < 16; off <<= 1) v += __shfl_xor(v, off, 32);
      l_r[r] = l_r[r] * al[r] + v;
    }
#pragma unroll
    for (int nt = 0; nt < 4; ++nt)
#pragma unroll
      for (int r = 0; r < 8; ++r) oacc[nt][r] *= al[r];

    // P -> LDS (per-wave private rows), re-read as A-frags
#pragma unroll
    for (int nt = 0; nt < 4; ++nt)
#pragma unroll
      for (int r = 0; r < 8; ++r)
        prow[(r + g * 8) * LDA + nt * 16 + ln] = f2bf(sacc[nt][r]);

#pragma unroll
    for (int kb = 0; kb < 2; ++kb) {
      FragAB pf;
      load_frag(pf, prow, LDA, kb * 32, lane);
#pragma unroll
      for (int nt = 0; nt < 4; ++nt) {
        FragAB vf;
        load_frag(vf, &Vs[(nt * 16) * LDA], LDA, kb * 32, lane);
        oacc[nt] = wmma_bf16(pf, vf, oacc[nt]);
      }
    }
  }

  float inv[8];
#pragma unroll
  for (int r = 0; r < 8; ++r) inv[r] = 1.0f / l_r[r];
#pragma unroll
  for (int nt = 0; nt < 4; ++nt)
#pragma unroll
    for (int r = 0; r < 8; ++r) {
      const int srow = q0 + wave * 16 + r + g * 8;
      const int col  = h * 64 + nt * 16 + ln;
      O[((size_t)b * S + srow) * (size_t)(H * 64) + col] = f2bf(oacc[nt][r] * inv[r]);
    }
}

// ---------------------------------------------------------------------------
// LayerNorm (D=1024, 256 threads/row) -> bf16
// ---------------------------------------------------------------------------
__global__ __launch_bounds__(256) void ln_kernel(
    const float* __restrict__ X, const float* __restrict__ sc,
    const float* __restrict__ bi, unsigned short* __restrict__ Y, int D)
{
  __shared__ float red[256];
  __shared__ float red2[256];
  const int row = blockIdx.x, tid = threadIdx.x;
  const float4 xv = *(const float4*)(X + (size_t)row * D + tid * 4);
  red[tid]  = xv.x + xv.y + xv.z + xv.w;
  red2[tid] = xv.x * xv.x + xv.y * xv.y + xv.z * xv.z + xv.w * xv.w;
  __syncthreads();
  for (int o = 128; o > 0; o >>= 1) {
    if (tid < o) { red[tid] += red[tid + o]; red2[tid] += red2[tid + o]; }
    __syncthreads();
  }
  const float mu  = red[0] / D;
  const float var = red2[0] / D - mu * mu;
  const float iv  = rsqrtf(var + 1e-6f);
  const int base  = tid * 4;
  Y[(size_t)row * D + base + 0] = f2bf((xv.x - mu) * iv * sc[base + 0] + bi[base + 0]);
  Y[(size_t)row * D + base + 1] = f2bf((xv.y - mu) * iv * sc[base + 1] + bi[base + 1]);
  Y[(size_t)row * D + base + 2] = f2bf((xv.z - mu) * iv * sc[base + 2] + bi[base + 2]);
  Y[(size_t)row * D + base + 3] = f2bf((xv.w - mu) * iv * sc[base + 3] + bi[base + 3]);
}

// ---------------------------------------------------------------------------
// Router: routing[t,e] = softmax_e( y[t,:] . W[:,e] ), E=8, one wave per row
// ---------------------------------------------------------------------------
__global__ __launch_bounds__(32) void router_kernel(
    const unsigned short* __restrict__ Y, const float* __restrict__ W,
    float* __restrict__ R, int D)
{
  const int row = blockIdx.x, lane = threadIdx.x;
  float a[8];
#pragma unroll
  for (int e = 0; e < 8; ++e) a[e] = 0.0f;
  for (int d = lane; d < D; d += 32) {
    const float yv = bf2f(Y[(size_t)row * D + d]);
    const float* wr = W + (size_t)d * 8;
#pragma unroll
    for (int e = 0; e < 8; ++e) a[e] += yv * wr[e];
  }
#pragma unroll
  for (int off = 16; off > 0; off >>= 1)
#pragma unroll
    for (int e = 0; e < 8; ++e) a[e] += __shfl_xor(a[e], off, 32);
  float mx = a[0];
#pragma unroll
  for (int e = 1; e < 8; ++e) mx = fmaxf(mx, a[e]);
  float sum = 0.0f;
#pragma unroll
  for (int e = 0; e < 8; ++e) { a[e] = __expf(a[e] - mx); sum += a[e]; }
  const float is = 1.0f / sum;
  if (lane == 0) {
#pragma unroll
    for (int e = 0; e < 8; ++e) R[(size_t)row * 8 + e] = a[e] * is;
  }
}

// ---------------------------------------------------------------------------
// Utility kernels
// ---------------------------------------------------------------------------
// dst[e][n][k] = bf16(src[e][k][n])    (weight transpose+convert)
__global__ void transpose_bf16_kernel(const float* __restrict__ src,
                                      unsigned short* __restrict__ dst,
                                      int K, int N, long total)
{
  const long kn = (long)K * N;
  for (long i = (long)blockIdx.x * blockDim.x + threadIdx.x; i < total;
       i += (long)gridDim.x * blockDim.x) {
    const long e  = i / kn;
    const long rp = i - e * kn;
    const int  n  = (int)(rp / K);
    const int  k  = (int)(rp - (long)n * K);
    dst[i] = f2bf(src[e * kn + (long)k * N + n]);
  }
}

__global__ void f32_to_bf16_kernel(const float* __restrict__ src,
                                   unsigned short* __restrict__ dst, long n)
{
  for (long i = (long)blockIdx.x * blockDim.x + threadIdx.x; i < n;
       i += (long)gridDim.x * blockDim.x) dst[i] = f2bf(src[i]);
}

__global__ void zero_f32_kernel(float* __restrict__ p, long n)
{
  for (long i = (long)blockIdx.x * blockDim.x + threadIdx.x; i < n;
       i += (long)gridDim.x * blockDim.x) p[i] = 0.0f;
}

// qkv bf16 [B,S,3,H,64] -> q,k [B*H,S,64], vt [B*H,64,S]
__global__ void qkv_reshape_kernel(const unsigned short* __restrict__ qkv,
                                   unsigned short* __restrict__ q,
                                   unsigned short* __restrict__ k,
                                   unsigned short* __restrict__ vt,
                                   int B, int S, int H, long total)
{
  for (long i = (long)blockIdx.x * blockDim.x + threadIdx.x; i < total;
       i += (long)gridDim.x * blockDim.x) {
    const int d = (int)(i & 63);
    long t = i >> 6;
    const int hh = (int)(t % H); t /= H;
    const int w  = (int)(t % 3); t /= 3;
    const int s  = (int)(t % S);
    const int b  = (int)(t / S);
    const unsigned short val = qkv[i];
    const long bh = (long)b * H + hh;
    if      (w == 0) q [(bh * S + s) * 64 + d] = val;
    else if (w == 1) k [(bh * S + s) * 64 + d] = val;
    else             vt[(bh * 64 + d) * S + s] = val;
  }
}

// ---------------------------------------------------------------------------
// Host orchestration
// ---------------------------------------------------------------------------
extern "C" void kernel_launch(void* const* d_in, const int* in_sizes, int n_in,
                              void* d_out, int out_size, void* d_ws, size_t ws_size,
                              hipStream_t stream)
{
  const int B = 2, S = 2048, Dm = 1024, H = 16, Dh = 64, E = 8, De = 1024;
  const int M = B * S;                 // 4096 tokens
  const int HD = H * Dh;               // 1024
  const int N3 = 3 * HD;               // 3072

  const float* x        = (const float*)d_in[0];
  const float* ln1_s    = (const float*)d_in[1];
  const float* ln1_b    = (const float*)d_in[2];
  const float* Wqkv     = (const float*)d_in[3];
  const float* bqkv     = (const float*)d_in[4];
  const float* Wo       = (const float*)d_in[5];
  const float* bo       = (const float*)d_in[6];
  const float* ln2_s    = (const float*)d_in[7];
  const float* ln2_b    = (const float*)d_in[8];
  const float* router_W = (const float*)d_in[9];
  const float* We       = (const float*)d_in[10];
  const float* be       = (const float*)d_in[11];
  const float* Wmo      = (const float*)d_in[12];
  const float* bmo      = (const float*)d_in[13];
  float* out = (float*)d_out;

  uint8_t* w8 = (uint8_t*)d_ws;
  size_t off = 0;
  auto AL = [&](size_t b) { size_t r = off; off += (b + 255) & ~(size_t)255; return r; };

  const size_t ofs_yb    = AL((size_t)M * Dm * 2);         // LN1 out bf16
  const size_t ofs_wqkvT = AL((size_t)Dm * N3 * 2);        // (reused: WoT@0, WmoT@+2MiB)
  const size_t ofs_qkvb  = AL((size_t)M * N3 * 2);         // (reused: comb f32@0, comb bf16@+16MiB)
  const size_t ofs_qb    = AL((size_t)B * H * S * Dh * 2); // (qb+kb reused as WeT)
  const size_t ofs_kb    = AL((size_t)B * H * S * Dh * 2);
  const size_t ofs_vtb   = AL((size_t)B * H * S * Dh * 2);
  const size_t ofs_ob    = AL((size_t)M * HD * 2);
  const size_t ofs_x1    = AL((size_t)M * Dm * 4);
  const size_t ofs_y2b   = AL((size_t)M * Dm * 2);
  const size_t ofs_rout  = AL((size_t)M * E * 4);
  (void)ws_size; (void)n_in; (void)in_sizes; (void)out_size;

  unsigned short* yb    = (unsigned short*)(w8 + ofs_yb);
  unsigned short* WqkvT = (unsigned short*)(w8 + ofs_wqkvT);
  unsigned short* WoT   = WqkvT;                                         // after GEMM1
  unsigned short* WmoT  = (unsigned short*)(w8 + ofs_wqkvT + (size_t)2 * 1024 * 1024);
  unsigned short* qkvb  = (unsigned short*)(w8 + ofs_qkvb);
  float*          comb  = (float*)(w8 + ofs_qkvb);                       // after reshape
  unsigned short* combb = (unsigned short*)(w8 + ofs_qkvb + (size_t)16 * 1024 * 1024);
  unsigned short* qb    = (unsigned short*)(w8 + ofs_qb);
  unsigned short* kb    = (unsigned short*)(w8 + ofs_kb);
  unsigned short* vtb   = (unsigned short*)(w8 + ofs_vtb);
  unsigned short* WeT   = (unsigned short*)(w8 + ofs_qb);                // after attention
  unsigned short* ob    = (unsigned short*)(w8 + ofs_ob);
  float*          x1    = (float*)(w8 + ofs_x1);
  unsigned short* y2b   = (unsigned short*)(w8 + ofs_y2b);
  float*          rout  = (float*)(w8 + ofs_rout);

  auto ew_blocks = [](long total) {
    long b = (total + 255) / 256; if (b > 16384) b = 16384; return (int)b;
  };

  // 1. Wqkv^T (bf16)
  {
    long tot = (long)Dm * N3;
    transpose_bf16_kernel<<<ew_blocks(tot), 256, 0, stream>>>(Wqkv, WqkvT, Dm, N3, tot);
  }
  // 2. LN1
  ln_kernel<<<M, 256, 0, stream>>>(x, ln1_s, ln1_b, yb, Dm);
  // 3. QKV = LN1(x) @ Wqkv + bqkv  -> bf16 [M, 3072]
  gemm_bf16_kernel<<<dim3(N3 / 128, M / 128), 256, 0, stream>>>(
      yb, WqkvT, bqkv, nullptr, nullptr, 0, (void*)qkvb, 1, 0, M, N3, Dm);
  // 4. reshape q/k/v^T
  {
    long tot = (long)M * N3;
    qkv_reshape_kernel<<<ew_blocks(tot), 256, 0, stream>>>(qkvb, qb, kb, vtb, B, S, H, tot);
  }
  // 5. Wo^T (reuses WqkvT memory; stream-ordered after GEMM1)
  {
    long tot = (long)HD * Dm;
    transpose_bf16_kernel<<<ew_blocks(tot), 256, 0, stream>>>(Wo, WoT, HD, Dm, tot);
  }
  // 6. flash attention -> o bf16 [M, 1024]
  attn_kernel<<<dim3(S / 64, B * H), 128, 0, stream>>>(qb, kb, vtb, ob, S, H, 0.125f);
  // 7. x1 = x + o @ Wo + bo
  gemm_bf16_kernel<<<dim3(Dm / 128, M / 128), 256, 0, stream>>>(
      ob, WoT, bo, x, nullptr, 0, (void*)x1, 0, 0, M, Dm, HD);
  // 8. LN2
  ln_kernel<<<M, 256, 0, stream>>>(x1, ln2_s, ln2_b, y2b, Dm);
  // 9. routing = softmax(y2 @ router_W)
  router_kernel<<<M, 32, 0, stream>>>(y2b, router_W, rout, Dm);
  // 10. We^T per expert (reuses q/k memory; after attention)
  {
    long tot = (long)E * Dm * De;
    transpose_bf16_kernel<<<ew_blocks(tot), 256, 0, stream>>>(We, WeT, Dm, De, tot);
  }
  // 11. Wmo^T
  {
    long tot = (long)De * Dm;
    transpose_bf16_kernel<<<ew_blocks(tot), 256, 0, stream>>>(Wmo, WmoT, De, Dm, tot);
  }
  // 12. combined = 0
  zero_f32_kernel<<<ew_blocks((long)M * De), 256, 0, stream>>>(comb, (long)M * De);
  // 13. combined += routing[:,e] * (y2 @ We[e] + be[e])   (8 stream-ordered GEMMs)
  for (int e = 0; e < E; ++e) {
    gemm_bf16_kernel<<<dim3(De / 128, M / 128), 256, 0, stream>>>(
        y2b, WeT + (size_t)e * De * Dm, be + (size_t)e * De, nullptr,
        rout + e, E, (void*)comb, 0, 1, M, De, Dm);
  }
  // 14. combined -> bf16
  f32_to_bf16_kernel<<<ew_blocks((long)M * De), 256, 0, stream>>>(comb, combb, (long)M * De);
  // 15. out = x1 + combined @ Wmo + bmo
  gemm_bf16_kernel<<<dim3(Dm / 128, M / 128), 256, 0, stream>>>(
      combb, WmoT, bmo, x1, nullptr, 0, (void*)out, 0, 0, M, Dm, De);
}
